// DeltaNet_31877247271516
// MI455X (gfx1250) — compile-verified, matrix operand order checked by reading, not run on hardware
//
#include <hip/hip_runtime.h>
#include <hip/hip_bf16.h>
#include <math.h>

#define DEV __device__ __forceinline__

typedef __attribute__((ext_vector_type(16))) _Float16 v16h;
typedef __attribute__((ext_vector_type(8)))  float    v8f;
typedef _Float16 h16;

constexpr int Bb = 2, Tt = 2048, Dd = 1024, Hh = 4, Pp = 6, CHc = 32;
constexpr int DKk = 256, DVv = 256;
constexpr int NCHUNK = Tt / CHc;           // 64
constexpr int GIW = Dd + Hh * Pp;          // 1048
constexpr float FLOORv = 0.05f;

// ---------------- WMMA fragment helpers (16x16x32 f16 -> f32 accum) --------
// A/B 16-bit fragment striping per CDNA5 ISA 7.12.2:
// lanes 0-15 = rows, K split: V0..V3 hold K[0..7]/[8..15], V4..V7 hold K[16..23]/[24..31]
DEV v16h frag_row(const h16* t, int ld, int r0, int k0, int lane) {
  // element [m][k] at t[(r0+m)*ld + (k0+k)]
  v16h f;
  int m = lane & 15, kg = lane >> 4;
#pragma unroll
  for (int j = 0; j < 8; ++j) {
    int kk = ((j < 4) ? 0 : 16) + kg * 8 + 2 * (j & 3);
    const h16* p = t + (size_t)(r0 + m) * ld + k0 + kk;
    f[2 * j] = p[0];
    f[2 * j + 1] = p[1];
  }
  return f;
}
DEV v16h frag_kmaj(const h16* t, int ld, int r0, int k0, int lane) {
  // element [m][k] at t[(k0+k)*ld + (r0+m)]
  v16h f;
  int m = lane & 15, kg = lane >> 4;
#pragma unroll
  for (int j = 0; j < 8; ++j) {
    int kk = ((j < 4) ? 0 : 16) + kg * 8 + 2 * (j & 3);
    f[2 * j]     = t[(size_t)(k0 + kk) * ld + r0 + m];
    f[2 * j + 1] = t[(size_t)(k0 + kk + 1) * ld + r0 + m];
  }
  return f;
}
DEV v8f wmma_f16(v16h a, v16h b, v8f c) {
  return __builtin_amdgcn_wmma_f32_16x16x32_f16(false, a, false, b, (short)0, c,
                                                false, false);
}
// C/D layout: VGPR r, lanes 0-15 -> M=r N=lane ; lanes 16-31 -> M=8+r N=lane-16

// ---------------- Generic GEMM: C = act(A(MxK) * W(NxK)^T + bias) ----------
// 64x128 block tile, 8 waves (2x4), each wave owns a 32x32 region (4 wmma/Kstep)
__global__ void __launch_bounds__(256)
gemm_kernel(const float* __restrict__ A, int lda, const float* __restrict__ W,
            int ldw, const float* __restrict__ bias, float* __restrict__ C,
            int ldc, int M, int N, int K, int act) {
  __shared__ h16 At[64][40];
  __shared__ h16 Wt[128][40];
  const int tid = threadIdx.x, lane = tid & 31, wid = tid >> 5;
  const int wm = wid >> 2, wn = wid & 3;
  const int bm = blockIdx.y * 64, bn = blockIdx.x * 128;
  const int r8 = tid >> 3, c4 = (tid & 7) * 4;
  v8f acc00 = {}, acc01 = {}, acc10 = {}, acc11 = {};
  const int ktn = (K + 31) >> 5;
  for (int kt = 0; kt < ktn; ++kt) {
    const int kb = kt << 5;
    const bool full = (bm + 64 <= M) && (bn + 128 <= N) && (kb + 32 <= K);
    if (full) {
      // branch-free interior path: float4 loads -> global_load_b128
#pragma unroll
      for (int i = 0; i < 2; ++i) {
        int r = r8 + 32 * i;
        const float4 f = *(const float4*)&A[(size_t)(bm + r) * lda + kb + c4];
        h16* d = &At[r][c4];
        d[0] = (h16)f.x; d[1] = (h16)f.y; d[2] = (h16)f.z; d[3] = (h16)f.w;
      }
#pragma unroll
      for (int i = 0; i < 4; ++i) {
        int r = r8 + 32 * i;
        const float4 f = *(const float4*)&W[(size_t)(bn + r) * ldw + kb + c4];
        h16* d = &Wt[r][c4];
        d[0] = (h16)f.x; d[1] = (h16)f.y; d[2] = (h16)f.z; d[3] = (h16)f.w;
      }
      if (kb + 64 <= K) {  // prefetch next K tile (global_prefetch_b8)
        __builtin_prefetch(&A[(size_t)(bm + r8) * lda + kb + 32 + c4], 0, 1);
        __builtin_prefetch(&W[(size_t)(bn + r8) * ldw + kb + 32 + c4], 0, 1);
      }
    } else {
      // guarded edge path (N=24 logits GEMM, K=1048 tail)
#pragma unroll
      for (int i = 0; i < 8; ++i) {
        int e = tid + i * 256, r = e >> 5, c = e & 31;
        int gm = bm + r, gk = kb + c;
        At[r][c] = (gm < M && gk < K) ? (h16)A[(size_t)gm * lda + gk] : (h16)0.f;
      }
#pragma unroll
      for (int i = 0; i < 16; ++i) {
        int e = tid + i * 256, r = e >> 5, c = e & 31;
        int gn = bn + r, gk = kb + c;
        Wt[r][c] = (gn < N && gk < K) ? (h16)W[(size_t)gn * ldw + gk] : (h16)0.f;
      }
    }
    __syncthreads();
    v16h a0 = frag_row(&At[0][0], 40, wm * 32, 0, lane);
    v16h a1 = frag_row(&At[0][0], 40, wm * 32 + 16, 0, lane);
    v16h b0 = frag_row(&Wt[0][0], 40, wn * 32, 0, lane);
    v16h b1 = frag_row(&Wt[0][0], 40, wn * 32 + 16, 0, lane);
    acc00 = wmma_f16(a0, b0, acc00);
    acc01 = wmma_f16(a0, b1, acc01);
    acc10 = wmma_f16(a1, b0, acc10);
    acc11 = wmma_f16(a1, b1, acc11);
    __syncthreads();
  }
  v8f* accs[4] = {&acc00, &acc01, &acc10, &acc11};
#pragma unroll
  for (int q = 0; q < 4; ++q) {
    int im = q >> 1, in_ = q & 1;
#pragma unroll
    for (int r = 0; r < 8; ++r) {
      int gm = bm + wm * 32 + im * 16 + r + 8 * (lane >> 4);
      int gn = bn + wn * 32 + in_ * 16 + (lane & 15);
      if (gm < M && gn < N) {
        float v = (*accs[q])[r] + (bias ? bias[gn] : 0.f);
        if (act == 1) v = 0.5f * v * (1.f + erff(v * 0.70710678118f));
        C[(size_t)gm * ldc + gn] = v;
      }
    }
  }
}

// ---------------- depthwise causal conv (+ optional SiLU) ------------------
__global__ void dwconv_kernel(const float* __restrict__ X,
                              const float* __restrict__ Wc,
                              float* __restrict__ Y, int Kw, int applySilu) {
  size_t idx = (size_t)blockIdx.x * blockDim.x + threadIdx.x;
  size_t total = (size_t)Bb * Tt * Dd;
  if (idx >= total) return;
  int c = (int)(idx & (Dd - 1));
  size_t bt = idx / Dd;
  int t = (int)(bt % Tt);
  float s = 0.f;
  for (int i = 0; i < Kw; ++i) {
    int tt = t - (Kw - 1) + i;
    if (tt >= 0) s += X[(bt - (size_t)(t - tt)) * Dd + c] * Wc[c * Kw + i];
  }
  if (applySilu) s = s / (1.f + expf(-s));
  Y[idx] = s;
}

// ---------------- beta = sigmoid(hidden @ b_proj^T) ------------------------
__global__ void beta_kernel(const float* __restrict__ hs,
                            const float* __restrict__ Wb,
                            float* __restrict__ beta) {
  int idx = blockIdx.x * blockDim.x + threadIdx.x;
  if (idx >= Bb * Tt * Hh) return;
  int h = idx & (Hh - 1);
  size_t bt = (size_t)idx >> 2;
  const float* x = hs + bt * Dd;
  const float* w = Wb + (size_t)h * Dd;
  float s = 0.f;
  for (int k = 0; k < Dd; ++k) s += x[k] * w[k];
  beta[idx] = 1.f / (1.f + expf(-s));
}

// ---------------- gi[:, :D] = hidden ---------------------------------------
__global__ void gi_copy_kernel(const float* __restrict__ hs,
                               float* __restrict__ gi) {
  size_t idx = (size_t)blockIdx.x * blockDim.x + threadIdx.x;
  if (idx >= (size_t)Bb * Tt * Dd) return;
  size_t bt = idx / Dd;
  int j = (int)(idx % Dd);
  gi[bt * GIW + j] = hs[idx];
}

// ---------------- stats = mean |branch| over DV -> gi tail -----------------
__global__ void stats_kernel(const float* __restrict__ b0,
                             const float* __restrict__ b1,
                             const float* __restrict__ b2,
                             const float* __restrict__ b3,
                             const float* __restrict__ b4,
                             const float* __restrict__ b5,
                             float* __restrict__ gi) {
  int wg = blockIdx.x * 8 + (threadIdx.x >> 5);
  int lane = threadIdx.x & 31;
  if (wg >= Bb * Tt * Hh * Pp) return;
  int p = wg % Pp;
  int h = (wg / Pp) % Hh;
  size_t bt = (size_t)wg / (Pp * Hh);
  const float* src = (p == 0) ? b0 : (p == 1) ? b1 : (p == 2) ? b2
                   : (p == 3) ? b3 : (p == 4) ? b4 : b5;
  src += bt * Dd + (size_t)h * DVv;
  float s = 0.f;
#pragma unroll
  for (int i = 0; i < 8; ++i) s += fabsf(src[lane + 32 * i]);
#pragma unroll
  for (int off = 16; off; off >>= 1) s += __shfl_xor(s, off, 32);
  if (lane == 0) gi[bt * GIW + Dd + h * Pp + p] = s * (1.f / 256.f);
}

// ---------------- chunked delta rule, LDS-resident 256x256 state -----------
__global__ void __launch_bounds__(256)
delta_kernel(const float* __restrict__ qc, const float* __restrict__ kc,
             const float* __restrict__ vc, const float* __restrict__ betap,
             float* __restrict__ dout) {
  extern __shared__ char smem[];
  float* uf  = (float*)smem;                 // 32x256 f32 (u then ui)
  float* Am  = uf + 32 * 256;                // 32x33 f32 (A / fwd-subst)
  h16* Sld   = (h16*)(Am + 32 * 33);         // 256x256 state (dk x dv)
  h16* qn    = Sld + 256 * 256;              // 32x256
  h16* kn    = qn + 32 * 256;
  h16* kbn   = kn + 32 * 256;
  h16* wn    = kbn + 32 * 256;               // holds -(inv@kb)
  h16* xb    = wn + 32 * 256;                // v*beta, then ui (f16)
  h16* invb  = xb + 32 * 256;                // 32x32
  h16* attb  = invb + 32 * 32;               // 32x32

  const int tid = threadIdx.x, lane = tid & 31, wid = tid >> 5;
  const int b = blockIdx.x >> 2, h = blockIdx.x & 3;

  for (int e = tid; e < 256 * 256; e += 256) Sld[e] = (h16)0.f;
  __syncthreads();

  for (int nc = 0; nc < NCHUNK; ++nc) {
    const int t0 = nc * CHc;
    // ---- stage 1: load chunk, l2norm q/k, apply beta ----
    for (int j = 0; j < 4; ++j) {
      int r = wid * 4 + j;
      int t = t0 + r;
      size_t rowb = ((size_t)(b * Tt + t)) * Dd + (size_t)h * DKk;
      float bet = betap[(size_t)(b * Tt + t) * Hh + h];
      float qv[8], kv[8], vv[8];
      float sq = 0.f, sk = 0.f;
#pragma unroll
      for (int i = 0; i < 8; ++i) {
        int d = lane + 32 * i;
        qv[i] = qc[rowb + d]; kv[i] = kc[rowb + d]; vv[i] = vc[rowb + d];
        sq += qv[i] * qv[i]; sk += kv[i] * kv[i];
      }
#pragma unroll
      for (int off = 16; off; off >>= 1) {
        sq += __shfl_xor(sq, off, 32);
        sk += __shfl_xor(sk, off, 32);
      }
      float rq = rsqrtf(sq + 1e-12f), rk = rsqrtf(sk + 1e-12f);
#pragma unroll
      for (int i = 0; i < 8; ++i) {
        int d = lane + 32 * i;
        float kq = kv[i] * rk;
        qn[r * 256 + d]  = (h16)(qv[i] * rq);
        kn[r * 256 + d]  = (h16)kq;
        kbn[r * 256 + d] = (h16)(kq * bet);
        xb[r * 256 + d]  = (h16)(vv[i] * bet);
      }
    }
    __syncthreads();
    // ---- stage 2: A = -(kb k^T), strict lower ----
    if (wid < 4) {
      int m0 = (wid >> 1) * 16, n0 = (wid & 1) * 16;
      v8f acc = {};
      for (int kk = 0; kk < 8; ++kk)
        acc = wmma_f16(frag_row(kbn, 256, m0, kk * 32, lane),
                       frag_row(kn, 256, n0, kk * 32, lane), acc);
#pragma unroll
      for (int r = 0; r < 8; ++r) {
        int m = m0 + r + 8 * (lane >> 4), n = n0 + (lane & 15);
        Am[m * 33 + n] = (m > n) ? -acc[r] : 0.f;
      }
    }
    __syncthreads();
    // ---- stage 3: forward substitution, inv = A + I ----
    for (int i = 1; i < 32; ++i) {
      float val = 0.f;
      if (tid < 32)
        for (int s = 0; s < i; ++s) val += Am[i * 33 + s] * Am[s * 33 + tid];
      __syncthreads();
      if (tid < 32) Am[i * 33 + tid] += val;
      __syncthreads();
    }
    for (int e = tid; e < 1024; e += 256) {
      int m = e >> 5, n = e & 31;
      invb[e] = (h16)(Am[m * 33 + n] + ((m == n) ? 1.f : 0.f));
    }
    __syncthreads();
    // ---- stage 4: u = inv@(v*beta) -> uf ; wn = -(inv@kb) ----
    for (int it = 0; it < 8; ++it) {
      int j = wid + 8 * it;           // 0..63
      int jj = j & 31;
      int m0 = (jj >> 4) * 16, n0 = (jj & 15) * 16;
      v8f acc = {};
      acc = wmma_f16(frag_row(invb, 32, m0, 0, lane),
                     frag_kmaj((j < 32) ? xb : kbn, 256, n0, 0, lane), acc);
#pragma unroll
      for (int r = 0; r < 8; ++r) {
        int m = m0 + r + 8 * (lane >> 4), n = n0 + (lane & 15);
        if (j < 32) uf[m * 256 + n] = acc[r];
        else        wn[m * 256 + n] = (h16)(-acc[r]);
      }
    }
    __syncthreads();
    // ---- stage 5: ui = u + (-w)@S -> uf (f32) and xb (f16) ----
    for (int it = 0; it < 4; ++it) {
      int j = wid + 8 * it;           // 0..31
      int m0 = (j >> 4) * 16, n0 = (j & 15) * 16;
      v8f acc;
#pragma unroll
      for (int r = 0; r < 8; ++r)
        acc[r] = uf[(m0 + r + 8 * (lane >> 4)) * 256 + n0 + (lane & 15)];
      for (int kk = 0; kk < 8; ++kk)
        acc = wmma_f16(frag_row(wn, 256, m0, kk * 32, lane),
                       frag_kmaj(Sld, 256, n0, kk * 32, lane), acc);
#pragma unroll
      for (int r = 0; r < 8; ++r) {
        int m = m0 + r + 8 * (lane >> 4), n = n0 + (lane & 15);
        uf[m * 256 + n] = acc[r];
        xb[m * 256 + n] = (h16)acc[r];
      }
    }
    __syncthreads();
    // ---- stage 6: attn = (q k^T) * lower-incl mask ----
    if (wid < 4) {
      int m0 = (wid >> 1) * 16, n0 = (wid & 1) * 16;
      v8f acc = {};
      for (int kk = 0; kk < 8; ++kk)
        acc = wmma_f16(frag_row(qn, 256, m0, kk * 32, lane),
                       frag_row(kn, 256, n0, kk * 32, lane), acc);
#pragma unroll
      for (int r = 0; r < 8; ++r) {
        int m = m0 + r + 8 * (lane >> 4), n = n0 + (lane & 15);
        attb[m * 32 + n] = (h16)((m >= n) ? acc[r] : 0.f);
      }
    }
    __syncthreads();
    // ---- stage 7: oi = q@S + attn@ui -> global ----
    for (int it = 0; it < 4; ++it) {
      int j = wid + 8 * it;
      int m0 = (j >> 4) * 16, n0 = (j & 15) * 16;
      v8f acc = {};
      for (int kk = 0; kk < 8; ++kk)
        acc = wmma_f16(frag_row(qn, 256, m0, kk * 32, lane),
                       frag_kmaj(Sld, 256, n0, kk * 32, lane), acc);
      acc = wmma_f16(frag_row(attb, 32, m0, 0, lane),
                     frag_kmaj(xb, 256, n0, 0, lane), acc);
#pragma unroll
      for (int r = 0; r < 8; ++r) {
        int m = m0 + r + 8 * (lane >> 4), n = n0 + (lane & 15);
        dout[((size_t)(b * Tt + t0 + m)) * Dd + (size_t)h * DVv + n] = acc[r];
      }
    }
    __syncthreads();
    // ---- stage 8: S += k^T @ ui ----
    for (int it = 0; it < 32; ++it) {
      int j = wid + 8 * it;           // 0..255
      int m0 = (j >> 4) * 16, n0 = (j & 15) * 16;
      v8f acc;
#pragma unroll
      for (int r = 0; r < 8; ++r)
        acc[r] = (float)Sld[(m0 + r + 8 * (lane >> 4)) * 256 + n0 + (lane & 15)];
      acc = wmma_f16(frag_kmaj(kn, 256, m0, 0, lane),
                     frag_kmaj(xb, 256, n0, 0, lane), acc);
#pragma unroll
      for (int r = 0; r < 8; ++r)
        Sld[(m0 + r + 8 * (lane >> 4)) * 256 + n0 + (lane & 15)] = (h16)acc[r];
    }
    __syncthreads();
  }
}

// ---------------- gate softmax + branch mixing + RMSNorm -------------------
__global__ void __launch_bounds__(256)
fuse_kernel(const float* __restrict__ b0, const float* __restrict__ b1,
            const float* __restrict__ b2, const float* __restrict__ b3,
            const float* __restrict__ b4, const float* __restrict__ b5,
            const float* __restrict__ gi, const float* __restrict__ logits,
            const float* __restrict__ alpha, const float* __restrict__ ltemp,
            const float* __restrict__ onw, float* __restrict__ out) {
  size_t bt = blockIdx.x;
  int tid = threadIdx.x;
  __shared__ float probs_s[Hh][Pp];
  __shared__ float red[256];
  if (tid < Hh) {
    int h = tid;
    float x = ltemp[h];
    float temp = fmaxf(x, 0.f) + log1pf(expf(-fabsf(x))) + 1e-4f; // softplus
    float l[Pp];
    float mx = -1e30f;
#pragma unroll
    for (int p = 0; p < Pp; ++p) {
      float st = gi[bt * GIW + Dd + h * Pp + p];
      l[p] = (logits[bt * (Hh * Pp) + h * Pp + p] + alpha[p] * st) / temp;
      mx = fmaxf(mx, l[p]);
    }
    float se = 0.f;
#pragma unroll
    for (int p = 0; p < Pp; ++p) { l[p] = expf(l[p] - mx); se += l[p]; }
#pragma unroll
    for (int p = 0; p < Pp; ++p) {
      float pr = l[p] / se * (1.f - FLOORv);
      if (p == Pp - 1) pr += FLOORv;
      probs_s[h][p] = pr;
    }
  }
  __syncthreads();
  int d = tid;
  for (int h = 0; h < Hh; ++h) {
    size_t base = bt * Dd + (size_t)h * DVv + d;
    float o = b0[base] * probs_s[h][0] + b1[base] * probs_s[h][1] +
              b2[base] * probs_s[h][2] + b3[base] * probs_s[h][3] +
              b4[base] * probs_s[h][4] + b5[base] * probs_s[h][5];
    red[tid] = o * o;
    __syncthreads();
    for (int s = 128; s; s >>= 1) {
      if (tid < s) red[tid] += red[tid + s];
      __syncthreads();
    }
    float ms = red[0] * (1.f / 256.f);
    __syncthreads();
    out[base] = o * rsqrtf(ms + 1e-5f) * onw[d];
  }
}

// ---------------------------------------------------------------------------
extern "C" void kernel_launch(void* const* d_in, const int* in_sizes, int n_in,
                              void* d_out, int out_size, void* d_ws,
                              size_t ws_size, hipStream_t stream) {
  const float* hs   = (const float*)d_in[0];
  const float* qw   = (const float*)d_in[1];
  const float* kw   = (const float*)d_in[2];
  const float* vw   = (const float*)d_in[3];
  const float* bw   = (const float*)d_in[4];
  const float* qcw  = (const float*)d_in[5];
  const float* kcw  = (const float*)d_in[6];
  const float* vcw  = (const float*)d_in[7];
  const float* f3w  = (const float*)d_in[8];
  const float* f7w  = (const float*)d_in[9];
  const float* f15w = (const float*)d_in[10];
  const float* f31w = (const float*)d_in[11];
  const float* gw1  = (const float*)d_in[12];
  const float* gb1  = (const float*)d_in[13];
  const float* gw2  = (const float*)d_in[14];
  const float* gb2  = (const float*)d_in[15];
  const float* glt  = (const float*)d_in[16];
  const float* alps = (const float*)d_in[17];
  const float* onw  = (const float*)d_in[18];
  const float* ow   = (const float*)d_in[19];
  float* out = (float*)d_out;

  const size_t U = (size_t)Bb * Tt * Dd; // 4096*1024
  float* ws = (float*)d_ws;
  float* qp  = ws;            // q projection (reused: fir3)
  float* kp  = qp + U;        // k projection (reused: fir7)
  float* vp  = kp + U;        // v projection (reused: fir15)
  float* qcv = vp + U;        // q after conv+silu (reused: fir31)
  float* kcv = qcv + U;       // k after conv+silu (reused: o_fused)
  float* vcv = kcv + U;       // v after conv+silu (branch p=4)
  float* dta = vcv + U;       // delta-rule output (branch p=5)
  float* h1  = dta + U;       // gate hidden, 2U
  float* gi  = h1 + 2 * U;    // gate input, B*T x GIW
  float* beta = gi + (size_t)Bb * Tt * 1056;
  float* lgt  = beta + (size_t)Bb * Tt * Hh;

  const int MT = Bb * Tt;                 // 4096
  dim3 blk(256);
  dim3 gN1024((1024 + 127) / 128, (MT + 63) / 64);
  int ebk = (int)((U + 255) / 256);

  // 1) q/k/v projections (no activation; conv+silu comes after)
  gemm_kernel<<<gN1024, blk, 0, stream>>>(hs, Dd, qw, Dd, nullptr, qp, Dd, MT, Dd, Dd, 0);
  gemm_kernel<<<gN1024, blk, 0, stream>>>(hs, Dd, kw, Dd, nullptr, kp, Dd, MT, Dd, Dd, 0);
  gemm_kernel<<<gN1024, blk, 0, stream>>>(hs, Dd, vw, Dd, nullptr, vp, Dd, MT, Dd, Dd, 0);
  // 2) beta
  beta_kernel<<<(MT * Hh + 255) / 256, blk, 0, stream>>>(hs, bw, beta);
  // 3) depthwise causal conv (k=4) + SiLU
  dwconv_kernel<<<ebk, blk, 0, stream>>>(qp, qcw, qcv, 4, 1);
  dwconv_kernel<<<ebk, blk, 0, stream>>>(kp, kcw, kcv, 4, 1);
  dwconv_kernel<<<ebk, blk, 0, stream>>>(vp, vcw, vcv, 4, 1);
  // 4) delta rule: 8 workgroups, ~248KB LDS-resident state each (320KB/WGP)
  const size_t DLDS = (32 * 256 + 32 * 33) * sizeof(float) +
                      (256 * 256 + 5 * 32 * 256 + 2 * 32 * 32) * sizeof(h16);
  delta_kernel<<<dim3(Bb * Hh), blk, DLDS, stream>>>(qcv, kcv, vcv, beta, dta);
  // 5) FIR branches on v (reuse proj buffers; qcv/kcv free after delta)
  dwconv_kernel<<<ebk, blk, 0, stream>>>(vcv, f3w, qp, 3, 0);
  dwconv_kernel<<<ebk, blk, 0, stream>>>(vcv, f7w, kp, 7, 0);
  dwconv_kernel<<<ebk, blk, 0, stream>>>(vcv, f15w, vp, 15, 0);
  dwconv_kernel<<<ebk, blk, 0, stream>>>(vcv, f31w, qcv, 31, 0);
  // 6) gate input = concat(hidden, stats)
  gi_copy_kernel<<<ebk, blk, 0, stream>>>(hs, gi);
  stats_kernel<<<(MT * Hh * Pp) / 8, blk, 0, stream>>>(qp, kp, vp, qcv, vcv, dta, gi);
  // 7) gate MLP (gelu, then linear to H*P logits with gb2 bias)
  dim3 gGate1((2048 + 127) / 128, (MT + 63) / 64);
  gemm_kernel<<<gGate1, blk, 0, stream>>>(gi, GIW, gw1, GIW, gb1, h1, 2048, MT, 2048, GIW, 1);
  dim3 gGate2(1, (MT + 63) / 64);
  gemm_kernel<<<gGate2, blk, 0, stream>>>(h1, 2048, gw2, 2048, gb2, lgt, Hh * Pp, MT, Hh * Pp, 2048, 0);
  // 8) softmax gate + mix branches + RMSNorm  (o_fused -> kcv)
  float* ofused = kcv;
  fuse_kernel<<<dim3(MT), blk, 0, stream>>>(qp, kp, vp, qcv, vcv, dta, gi, lgt,
                                            alps, glt, onw, ofused);
  // 9) output projection
  gemm_kernel<<<gN1024, blk, 0, stream>>>(ofused, Dd, ow, Dd, nullptr, out, Dd, MT, Dd, Dd, 0);
}
